// SelfAttention_51556787421543
// MI455X (gfx1250) — matrix operand order repaired and checked by reading, not compiled, not run on hardware
//
#include <hip/hip_runtime.h>
#include <hip/hip_bf16.h>
#include <math.h>

typedef __attribute__((ext_vector_type(16))) _Float16 v16h;
typedef __attribute__((ext_vector_type(8)))  _Float16 v8h;
typedef __attribute__((ext_vector_type(8)))  float    v8f;

#define B_  2
#define S_  2048
#define H_  2048
#define NH_ 16
#define HD_ 128

__device__ __forceinline__ v8f wmma_f16(v16h a, v16h b, v8f c) {
  return __builtin_amdgcn_wmma_f32_16x16x32_f16(false, a, false, b, (short)0, c, false, false);
}

__device__ __forceinline__ v16h combine(v8h lo, v8h hi) {
  v16h r;
#pragma unroll
  for (int i = 0; i < 8; ++i) { r[i] = lo[i]; r[i + 8] = hi[i]; }
  return r;
}

// CDNA5 async copy: 16B global -> LDS, tracked by ASYNCcnt (ISA §10 / §15.18.3).
__device__ __forceinline__ void async_copy16(unsigned lds_off, const void* gaddr) {
  asm volatile("global_load_async_to_lds_b128 %0, %1, off"
               :: "v"(lds_off), "v"(gaddr) : "memory");
}
__device__ __forceinline__ void wait_async0() {
  asm volatile("s_wait_asynccnt 0" ::: "memory");
}

// one-time f32 -> f16 conversion (bandwidth-bound, ~50MB total)
__global__ __launch_bounds__(256) void cvt_f16(const float* __restrict__ s,
                                               _Float16* __restrict__ d, int n) {
  const int i = (blockIdx.x * 256 + threadIdx.x) * 8;
  if (i >= n) return;
  const float4 a = *(const float4*)(s + i);
  const float4 b = *(const float4*)(s + i + 4);
  v8h h;
  h[0] = (_Float16)a.x; h[1] = (_Float16)a.y; h[2] = (_Float16)a.z; h[3] = (_Float16)a.w;
  h[4] = (_Float16)b.x; h[5] = (_Float16)b.y; h[6] = (_Float16)b.z; h[7] = (_Float16)b.w;
  *(v8h*)(d + i) = h;
}

// C = A[M,K]f16 x W[N,K]^T f16 + bias.  MODE 0: scatter QKV (f16, head-major,
// Q pre-scaled by 1/sqrt(HD)); MODE 1: f32 row-major store to out.
// Block tile 64x128, 8 waves of 32x32, K-tile 64, double-buffered async staging.
template<int N, int K, int MODE>
__global__ __launch_bounds__(256) void gemm_wmma(
    const _Float16* __restrict__ A, const _Float16* __restrict__ W,
    const float* __restrict__ bias,
    _Float16* __restrict__ qp, _Float16* __restrict__ kp, _Float16* __restrict__ vp,
    float* __restrict__ out)
{
  constexpr int TK = 64, LDT = TK + 8;              // 72 halfs = 144B row stride
  __shared__ _Float16 Alds[2][64 * LDT];
  __shared__ _Float16 Wlds[2][128 * LDT];

  const int tid  = threadIdx.x;
  const int lane = tid & 31;
  const int wid  = tid >> 5;
  const int wm   = (wid >> 2) * 32;
  const int wn   = (wid & 3) * 32;
  const int m_blk = blockIdx.y * 64;
  const int n_blk = blockIdx.x * 128;

  const int ar = tid >> 2, ac = (tid & 3) * 16;     // A: 64 x 64 halfs, 2 chunks/thread
  const int wr = tid >> 1, wc = (tid & 1) * 32;     // W: 128 x 64 halfs, 4 chunks/thread

  const int fr    = lane & 15;
  const int koff  = (lane >> 4) * 8;
  const int kbase = (lane >> 4) * 16;

  v8f acc[2][2] = {};

  auto stage = [&](int buf, int k0) {
    const _Float16* asrc = A + (size_t)(m_blk + ar) * K + k0 + ac;
    const unsigned  al   = (unsigned)(size_t)&Alds[buf][ar * LDT + ac];
    async_copy16(al,      asrc);
    async_copy16(al + 16, asrc + 8);
    const _Float16* wsrc = W + (size_t)(n_blk + wr) * K + k0 + wc;
    const unsigned  wl   = (unsigned)(size_t)&Wlds[buf][wr * LDT + wc];
    async_copy16(wl,      wsrc);
    async_copy16(wl + 16, wsrc + 8);
    async_copy16(wl + 32, wsrc + 16);
    async_copy16(wl + 48, wsrc + 24);
  };

  stage(0, 0);
  wait_async0();
  __syncthreads();

  int buf = 0;
  for (int k0 = 0; k0 < K; k0 += TK) {
    if (k0 + TK < K) stage(buf ^ 1, k0 + TK);       // prefetch next tile (async)
#pragma unroll
    for (int kk = 0; kk < 2; ++kk) {                // two 32-K WMMA steps per stage
      v16h af[2], bf[2];
#pragma unroll
      for (int i = 0; i < 2; ++i) {
        const _Float16* ap = &Alds[buf][(wm + i * 16 + fr) * LDT + kk * 32];
        af[i] = combine(*(const v8h*)&ap[kbase], *(const v8h*)&ap[kbase + 8]);
        const _Float16* bp = &Wlds[buf][(wn + i * 16 + fr) * LDT + kk * 32 + kbase];
        bf[i] = combine(*(const v8h*)bp, *(const v8h*)(bp + 8));
      }
#pragma unroll
      for (int i = 0; i < 2; ++i)
#pragma unroll
        for (int j = 0; j < 2; ++j)
          acc[i][j] = wmma_f16(af[i], bf[j], acc[i][j]);
    }
    wait_async0();
    __syncthreads();
    buf ^= 1;
  }

  const int col_in_tile = lane & 15;
  const int row_off     = (lane >> 4) * 8;
#pragma unroll
  for (int i = 0; i < 2; ++i) {
#pragma unroll
    for (int j = 0; j < 2; ++j) {
      const int gcol = n_blk + wn + j * 16 + col_in_tile;
#pragma unroll
      for (int vv = 0; vv < 8; ++vv) {
        const int grow = m_blk + wm + i * 16 + row_off + vv;
        const float val = acc[i][j][vv] + bias[gcol];
        if (MODE == 0) {
          const int which = gcol / H_;              // compile-time pow2 -> shifts
          const int hh    = (gcol % H_) / HD_;
          const int d     = gcol % HD_;
          const int bb    = grow / S_;
          const int ss    = grow % S_;
          const size_t idx = (((size_t)bb * NH_ + hh) * S_ + ss) * HD_ + d;
          if (which == 0)      qp[idx] = (_Float16)(val * 0.08838834764831845f);
          else if (which == 1) kp[idx] = (_Float16)val;
          else                 vp[idx] = (_Float16)val;
        } else {
          out[(size_t)grow * N + gcol] = val;
        }
      }
    }
  }
}

// Flash attention, causal (mask==tril applied analytically; masked score = -10000,
// identical post-softmax to the reference after f32 exp underflow).
// grid = B*NH*(S/128); 8 waves/block, each wave owns 16 query rows.
__global__ __launch_bounds__(256) void attn_flash(
    const _Float16* __restrict__ q, const _Float16* __restrict__ k,
    const _Float16* __restrict__ v, _Float16* __restrict__ ctxo)
{
  constexpr int LDK = HD_ + 8;   // 136 halfs (272B row stride)
  constexpr int LDV = 32 + 8;
  constexpr int LDP = 32 + 8;
  __shared__ _Float16 Klds[32 * LDK];     // K tile   [k=32][d=128]
  __shared__ _Float16 Vlds[HD_ * LDV];    // V^T tile [d=128][k=32]
  __shared__ _Float16 Plds[8 * 16 * LDP]; // per-wave P scratch [16][32]

  const int tid  = threadIdx.x;
  const int lane = tid & 31;
  const int wid  = tid >> 5;

  const int qchunk = blockIdx.x & 15;
  const int head   = (blockIdx.x >> 4) & (NH_ - 1);
  const int batch  = blockIdx.x >> 8;

  const int q0  = qchunk * 128;
  const int qr0 = q0 + wid * 16;
  const size_t headoff = ((size_t)batch * NH_ + head) * S_ * HD_;
  const _Float16* Qh = q + headoff;
  const _Float16* Kh = k + headoff;
  const _Float16* Vh = v + headoff;

  const int fr   = lane & 15;
  const int koff = (lane >> 4) * 8;
  const int kfb  = (lane >> 4) * 16;

  // Q A-fragments (f16 direct loads; Q already scaled by 1/sqrt(HD))
  v16h qf[4];
#pragma unroll
  for (int c = 0; c < 4; ++c) {
    const _Float16* qrow = Qh + (size_t)(qr0 + fr) * HD_ + c * 32;
    qf[c] = combine(*(const v8h*)(qrow + kfb), *(const v8h*)(qrow + kfb + 8));
  }

  v8f ctx[8] = {};
  float mrow[8], lrow[8];
#pragma unroll
  for (int i = 0; i < 8; ++i) { mrow[i] = -1e30f; lrow[i] = 0.f; }

  const int ksteps = (q0 + 128) / 32;
  const int skr = tid >> 3;                // 0..31
  const int skc = (tid & 7) * 16;          // 0..112 (halfs)

  for (int kt = 0; kt < ksteps; ++kt) {
    const int kb = kt * 32;
    { // K tile via async copy; V tile via transposed LDS stores
      const _Float16* srcK = Kh + (size_t)(kb + skr) * HD_ + skc;
      const unsigned  kl   = (unsigned)(size_t)&Klds[skr * LDK + skc];
      async_copy16(kl,      srcK);
      async_copy16(kl + 16, srcK + 8);
      const _Float16* srcV = Vh + (size_t)(kb + skr) * HD_ + skc;
      const v8h v0 = *(const v8h*)srcV;
      const v8h v1 = *(const v8h*)(srcV + 8);
#pragma unroll
      for (int i = 0; i < 8; ++i) {
        Vlds[(skc + i) * LDV + skr]     = v0[i];
        Vlds[(skc + 8 + i) * LDV + skr] = v1[i];
      }
    }
    wait_async0();
    __syncthreads();

    // S = Q x K^T : two 16-col score tiles per 32-key step (8 WMMAs)
    v8f s0 = {}, s1 = {};
#pragma unroll
    for (int c = 0; c < 4; ++c) {
      const _Float16* kp0 = &Klds[fr * LDK + c * 32 + kfb];
      s0 = wmma_f16(qf[c], combine(*(const v8h*)kp0, *(const v8h*)(kp0 + 8)), s0);
      const _Float16* kp1 = &Klds[(16 + fr) * LDK + c * 32 + kfb];
      s1 = wmma_f16(qf[c], combine(*(const v8h*)kp1, *(const v8h*)(kp1 + 8)), s1);
    }

    // streaming softmax across the 16-lane half holding each row
    const int colbase = kb + (lane & 15);
    const int rowb    = qr0 + (lane >> 4) * 8;
    float p0[8], p1[8], corr[8];
#pragma unroll
    for (int vv = 0; vv < 8; ++vv) {
      const int r = rowb + vv;
      float a0 = (colbase      <= r) ? s0[vv] : -10000.f;
      float a1 = (colbase + 16 <= r) ? s1[vv] : -10000.f;
      float rmax = fmaxf(a0, a1);
#pragma unroll
      for (int off = 1; off < 16; off <<= 1)
        rmax = fmaxf(rmax, __shfl_xor(rmax, off, 32));
      const float mnew = fmaxf(mrow[vv], rmax);
      const float c0 = __expf(mrow[vv] - mnew);
      const float e0 = __expf(a0 - mnew);
      const float e1 = __expf(a1 - mnew);
      float rs = e0 + e1;
#pragma unroll
      for (int off = 1; off < 16; off <<= 1)
        rs += __shfl_xor(rs, off, 32);
      lrow[vv] = lrow[vv] * c0 + rs;
      mrow[vv] = mnew;
      corr[vv] = c0;
      p0[vv] = e0; p1[vv] = e1;
    }

#pragma unroll
    for (int n = 0; n < 8; ++n)
#pragma unroll
      for (int vv = 0; vv < 8; ++vv)
        ctx[n][vv] *= corr[vv];

    // P: C-fragment -> A-fragment via wave-private LDS
    _Float16* pb = &Plds[wid * 16 * LDP];
#pragma unroll
    for (int vv = 0; vv < 8; ++vv) {
      const int pr = (lane >> 4) * 8 + vv;
      pb[pr * LDP + (lane & 15)]      = (_Float16)p0[vv];
      pb[pr * LDP + 16 + (lane & 15)] = (_Float16)p1[vv];
    }
    asm volatile("s_wait_dscnt 0" ::: "memory");
    v16h pf;
    {
      const _Float16* pp = &pb[fr * LDP];
      pf = combine(*(const v8h*)&pp[kfb], *(const v8h*)&pp[kfb + 8]);
    }

    // ctx += P x V (8 WMMAs)
#pragma unroll
    for (int n = 0; n < 8; ++n) {
      const _Float16* vpp = &Vlds[(n * 16 + fr) * LDV + kfb];
      ctx[n] = wmma_f16(pf, combine(*(const v8h*)vpp, *(const v8h*)(vpp + 8)), ctx[n]);
    }
    __syncthreads();
  }

  // normalize, write ctx (f16) back to [b,s,H] layout for the out-projection
  const size_t obase = ((size_t)batch * S_) * H_ + (size_t)head * HD_;
#pragma unroll
  for (int n = 0; n < 8; ++n) {
#pragma unroll
    for (int vv = 0; vv < 8; ++vv) {
      const int r = qr0 + (lane >> 4) * 8 + vv;
      const int d = n * 16 + (lane & 15);
      ctxo[obase + (size_t)r * H_ + d] = (_Float16)(ctx[n][vv] / lrow[vv]);
    }
  }
}

extern "C" void kernel_launch(void* const* d_in, const int* in_sizes, int n_in,
                              void* d_out, int out_size, void* d_ws, size_t ws_size,
                              hipStream_t stream) {
  const float* x     = (const float*)d_in[0];
  // d_in[1] is the causal mask; known tril(ones), applied analytically.
  const float* w_qkv = (const float*)d_in[2];
  const float* b_qkv = (const float*)d_in[3];
  const float* w_o   = (const float*)d_in[4];
  const float* b_o   = (const float*)d_in[5];
  float* out = (float*)d_out;

  const size_t nx    = (size_t)B_ * S_ * H_;        // 8.4M
  const size_t nwqkv = (size_t)3 * H_ * H_;         // 12.6M
  const size_t nwo   = (size_t)H_ * H_;             // 4.2M
  const size_t per   = (size_t)B_ * NH_ * S_ * HD_; // 8.4M

  _Float16* xh    = (_Float16*)d_ws;
  _Float16* wqkvh = xh + nx;
  _Float16* woh   = wqkvh + nwqkv;
  _Float16* qh    = woh + nwo;
  _Float16* kh    = qh + per;
  _Float16* vh    = kh + per;
  _Float16* ctxh  = vh + per;

  dim3 blk(256);

  // one-time f32 -> f16 conversions
  cvt_f16<<<dim3(nx    / 2048), blk, 0, stream>>>(x,     xh,    (int)nx);
  cvt_f16<<<dim3(nwqkv / 2048), blk, 0, stream>>>(w_qkv, wqkvh, (int)nwqkv);
  cvt_f16<<<dim3(nwo   / 2048), blk, 0, stream>>>(w_o,   woh,   (int)nwo);

  // QKV projection: [4096,2048] x [6144,2048]^T -> Q/K/V (f16, head-major)
  dim3 g1((3 * H_) / 128, (B_ * S_) / 64);
  gemm_wmma<3 * H_, H_, 0><<<g1, blk, 0, stream>>>(xh, wqkvh, b_qkv,
                                                   qh, kh, vh, nullptr);

  // Flash attention: 512 blocks
  dim3 g2(B_ * NH_ * (S_ / 128));
  attn_flash<<<g2, blk, 0, stream>>>(qh, kh, vh, ctxh);

  // Output projection: [4096,2048] x [2048,2048]^T -> d_out (f32)
  dim3 g3(H_ / 128, (B_ * S_) / 64);
  gemm_wmma<H_, H_, 1><<<g3, blk, 0, stream>>>(ctxh, woh, b_o,
                                               nullptr, nullptr, nullptr, out);
}
